// IntraSA_78907139162132
// MI455X (gfx1250) — matrix operand order repaired and checked by reading, not compiled, not run on hardware
//
#include <hip/hip_runtime.h>

#define HEADS    5
#define CHANNELS 320
#define SPLIT    160
#define HC       32
#define HWN      65536          // H*W
#define PTOT     131072         // B*H*W
#define LN_EPS   1e-5f

typedef __attribute__((ext_vector_type(16))) __bf16 v16bf;
typedef __attribute__((ext_vector_type(8)))  __bf16 v8bf;
typedef __attribute__((ext_vector_type(4)))  __bf16 v4bf;
typedef __attribute__((ext_vector_type(8)))  float  v8f;
typedef __attribute__((ext_vector_type(4)))  int    v4i;

#if __has_builtin(__builtin_amdgcn_global_load_async_to_lds_b128)
#define USE_ASYNC_LDS 1
typedef __attribute__((address_space(1))) v4i* gbl_v4i_ptr;
typedef __attribute__((address_space(3))) v4i* lds_v4i_ptr;
#else
#define USE_ASYNC_LDS 0
#endif

__device__ __forceinline__ void wait_async0() {
#if USE_ASYNC_LDS
#if __has_builtin(__builtin_amdgcn_s_wait_asynccnt)
    __builtin_amdgcn_s_wait_asynccnt(0);
#else
    asm volatile("s_wait_asynccnt 0" ::: "memory");
#endif
#endif
}

// Assemble a v16bf fragment from two contiguous 16B chunks (LDS or global).
__device__ __forceinline__ v16bf frag16(const __bf16* p0, const __bf16* p1) {
    v8bf x = *(const v8bf*)p0;
    v8bf y = *(const v8bf*)p1;
    return __builtin_shufflevector(x, y, 0,1,2,3,4,5,6,7,8,9,10,11,12,13,14,15);
}

// ---------------------------------------------------------------------------
// Kernel 1: channel LayerNorm; x (2,320,256,256) f32 -> xn bf16 [ch][pos]
// ---------------------------------------------------------------------------
__global__ void ln_kernel(const float* __restrict__ x,
                          const float* __restrict__ ln_w,
                          const float* __restrict__ ln_b,
                          __bf16* __restrict__ xn) {
    int g   = blockIdx.x * blockDim.x + threadIdx.x;   // 0..PTOT-1
    int b   = g >> 16;
    int pos = g & (HWN - 1);
    const float* xb = x + (size_t)b * CHANNELS * HWN + pos;
    float s = 0.f, s2 = 0.f;
    for (int ch = 0; ch < CHANNELS; ++ch) {
        float v = xb[(size_t)ch * HWN];
        s += v; s2 += v * v;
    }
    float mu   = s * (1.0f / CHANNELS);
    float var  = s2 * (1.0f / CHANNELS) - mu * mu;
    float rstd = rsqrtf(var + LN_EPS);
    for (int ch = 0; ch < CHANNELS; ++ch) {
        float v = xb[(size_t)ch * HWN];
        float y = (v - mu) * rstd * ln_w[ch] + ln_b[ch];
        xn[(size_t)ch * PTOT + g] = (__bf16)y;   // channel-major, coalesced
    }
}

// ---------------------------------------------------------------------------
// Kernel 2: 1x1 conv GEMM: xc[p][o] = sum_c xn[c][p] * w[o][c] + bias[o]
// block = 128 thr (4 waves), tile M=16 x N=64, K-step 32
// ---------------------------------------------------------------------------
__global__ void conv_gemm_kernel(const __bf16* __restrict__ xn,   // [320][PTOT]
                                 const float*  __restrict__ w,    // [320][320]
                                 const float*  __restrict__ bias, // [320]
                                 __bf16* __restrict__ xc) {       // [PTOT][320]
    __shared__ __align__(16) __bf16 As[16][40];   // [m][k]
    __shared__ __align__(16) __bf16 Bt[64][40];   // [n][k]  (B transposed)
    int tid  = threadIdx.x;
    int lane = tid & 31;
    int wv   = tid >> 5;                     // 0..3
    int m0   = blockIdx.x * 16;
    int n0   = blockIdx.y * 64;
    v8f acc = {0.f,0.f,0.f,0.f,0.f,0.f,0.f,0.f};

    for (int k0 = 0; k0 < CHANNELS; k0 += 32) {
        // A tile 16x32 from channel-major xn: 16B vector loads, b16 scatter
        if (tid < 64) {
            int k = tid >> 1, mh = (tid & 1) * 8;
            v8bf g = *(const v8bf*)&xn[(size_t)(k0 + k) * PTOT + m0 + mh];
            for (int j = 0; j < 8; ++j) As[mh + j][k] = g[j];
        }
        // B tile transposed: Bt[n][k] = w[n0+n][k0+k]; float4 loads
        for (int e = tid; e < 512; e += 128) {
            int n = e >> 3, k4 = (e & 7) * 4;
            float4 f = *(const float4*)&w[(size_t)(n0 + n) * CHANNELS + k0 + k4];
            v4bf o; o[0] = (__bf16)f.x; o[1] = (__bf16)f.y;
            o[2] = (__bf16)f.z; o[3] = (__bf16)f.w;
            *(v4bf*)&Bt[n][k4] = o;
        }
        if (k0 + 32 < CHANNELS)              // -> global_prefetch_b8
            __builtin_prefetch(&w[(size_t)(n0 + (tid >> 2)) * CHANNELS + k0 + 32], 0, 1);
        __syncthreads();

        int m  = lane & 15, hi = lane >> 4;
        v16bf a  = frag16(&As[m][hi ? 16 : 0], &As[m][hi ? 24 : 8]);
        int nn = (wv << 4) + m;
        v16bf bb = frag16(&Bt[nn][hi ? 16 : 0], &Bt[nn][(hi ? 16 : 0) + 8]);
        acc = __builtin_amdgcn_wmma_f32_16x16x32_bf16(false, a, false, bb,
                                                      (short)0, acc, false, false);
        __syncthreads();
    }
    int n  = n0 + (wv << 4) + (lane & 15);
    int hi = lane >> 4;
    float bsv = bias[n];
    for (int r = 0; r < 8; ++r) {
        int m = m0 + r + 8 * hi;
        xc[(size_t)m * CHANNELS + n] = (__bf16)(acc[r] + bsv);
    }
}

// ---------------------------------------------------------------------------
// Kernel 3: QKV projection GEMM: out[p][o] = sum_c xc[p][c_off+c]*w[o][c]+b[o]
// block = 160 thr (5 waves), tile M=16 x N=80, K=160.
// A tile is a straight copy -> async global->LDS engine.
// ---------------------------------------------------------------------------
__global__ void proj_gemm_kernel(const __bf16* __restrict__ xc,   // [PTOT][320]
                                 const float*  __restrict__ w,    // [160][160]
                                 const float*  __restrict__ bias, // [160]
                                 __bf16* __restrict__ out,        // [PTOT][160]
                                 int c_off) {
    __shared__ __align__(16) __bf16 As[16][40];
    __shared__ __align__(16) __bf16 Bt[80][40];
    int tid  = threadIdx.x;                  // 0..159
    int lane = tid & 31;
    int wv   = tid >> 5;                     // 0..4
    int m0   = blockIdx.x * 16;
    int n0   = blockIdx.y * 80;
    v8f acc = {0.f,0.f,0.f,0.f,0.f,0.f,0.f,0.f};

    for (int k0 = 0; k0 < SPLIT; k0 += 32) {
        // A tile: contiguous rows of xc -> async DMA to LDS (issued first)
        if (tid < 64) {
            int m = tid >> 2, c8 = (tid & 3) * 8;
            const __bf16* gp = &xc[(size_t)(m0 + m) * CHANNELS + c_off + k0 + c8];
#if USE_ASYNC_LDS
            __builtin_amdgcn_global_load_async_to_lds_b128(
                (gbl_v4i_ptr)const_cast<__bf16*>(gp), (lds_v4i_ptr)&As[m][c8], 0, 0);
#else
            *(v8bf*)&As[m][c8] = *(const v8bf*)gp;
#endif
        }
        // B tile transposed: Bt[n][k] = w[n0+n][k0+k]  (overlaps the async DMA)
        for (int e = tid; e < 640; e += 160) {
            int n = e >> 3, k4 = (e & 7) * 4;
            float4 f = *(const float4*)&w[(size_t)(n0 + n) * SPLIT + k0 + k4];
            v4bf o; o[0] = (__bf16)f.x; o[1] = (__bf16)f.y;
            o[2] = (__bf16)f.z; o[3] = (__bf16)f.w;
            *(v4bf*)&Bt[n][k4] = o;
        }
        wait_async0();
        __syncthreads();

        int m  = lane & 15, hi = lane >> 4;
        v16bf a  = frag16(&As[m][hi ? 16 : 0], &As[m][hi ? 24 : 8]);
        int nn = (wv << 4) + m;
        v16bf bb = frag16(&Bt[nn][hi ? 16 : 0], &Bt[nn][(hi ? 16 : 0) + 8]);
        acc = __builtin_amdgcn_wmma_f32_16x16x32_bf16(false, a, false, bb,
                                                      (short)0, acc, false, false);
        __syncthreads();
    }
    int n  = n0 + (wv << 4) + (lane & 15);
    int hi = lane >> 4;
    float bsv = bias[n];
    for (int r = 0; r < 8; ++r) {
        int m = m0 + r + 8 * hi;
        out[(size_t)m * SPLIT + n] = (__bf16)(acc[r] + bsv);
    }
}

// ---------------------------------------------------------------------------
// Kernel 4: axial attention, flash-style online softmax.
// grid = (512 seqs, 5 heads, 2 q-halves), block = 256 thr (8 waves).
// Double-buffered K/V chunks: async K DMA for chunk c+1 overlaps the
// WMMA + softmax work on chunk c; ASYNCcnt waited only at iteration end.
// ---------------------------------------------------------------------------
__global__ void axial_attn_kernel(const __bf16* __restrict__ qb,
                                  const __bf16* __restrict__ kb,
                                  const __bf16* __restrict__ vb,
                                  float* __restrict__ out,
                                  int dir) {
    __shared__ __align__(16) __bf16 Ks[2][32][40];    // [buf][tok][ch]
    __shared__ __align__(16) __bf16 Vt[2][32][40];    // [buf][ch][tok]
    __shared__ __align__(16) __bf16 Ps[8][16][40];    // per-wave P tile [m][k]
    __shared__ float  Ob[8][32][17];

    int tid  = threadIdx.x, lane = tid & 31, wv = tid >> 5;  // wv 0..7
    int seq  = blockIdx.x;                   // 0..511
    int head = blockIdx.y;
    int qt0  = blockIdx.z * 128 + wv * 16;   // first query token of this wave
    int b    = seq >> 8;
    int so   = seq & 255;                    // h (dir0) or w (dir1)
    int rowbase = b * HWN + (dir ? so : so * 256);
    int stride  = dir ? 256 : 1;
    int choff   = head * HC;
    const float scale = 0.17677669529663687f; // 1/sqrt(32)

    // Q fragment: two contiguous 16B global loads per lane
    int m  = lane & 15, hi = lane >> 4;
    const __bf16* qr = qb + (size_t)(rowbase + (qt0 + m) * stride) * SPLIT + choff;
    v16bf aq = frag16(qr + (hi ? 16 : 0), qr + (hi ? 24 : 8));

    // stage one 32-token K/V chunk into buffer bi
    auto stage = [&](int c0, int bi) {
        int e  = tid & 127;
        int t  = e >> 2, c8 = (e & 3) * 8;
        size_t roff = (size_t)(rowbase + (c0 + t) * stride) * SPLIT + choff + c8;
#if USE_ASYNC_LDS
        if (tid < 128) {
            __builtin_amdgcn_global_load_async_to_lds_b128(
                (gbl_v4i_ptr)const_cast<__bf16*>(kb + roff),
                (lds_v4i_ptr)&Ks[bi][t][c8], 0, 0);
        } else {
            v8bf g = *(const v8bf*)&vb[roff];
            for (int j = 0; j < 8; ++j) Vt[bi][c8 + j][t] = g[j];
        }
#else
        if (tid < 128) {
            *(v8bf*)&Ks[bi][t][c8] = *(const v8bf*)&kb[roff];
        } else {
            v8bf g = *(const v8bf*)&vb[roff];
            for (int j = 0; j < 8; ++j) Vt[bi][c8 + j][t] = g[j];
        }
#endif
    };

    float mrow[8], lrow[8];
    v8f o0 = {0.f,0.f,0.f,0.f,0.f,0.f,0.f,0.f};
    v8f o1 = {0.f,0.f,0.f,0.f,0.f,0.f,0.f,0.f};
    for (int r = 0; r < 8; ++r) { mrow[r] = -1e30f; lrow[r] = 0.f; }

    stage(0, 0);
    wait_async0();
    __syncthreads();

    int bi = 0;
    for (int c0 = 0; c0 < 256; c0 += 32, bi ^= 1) {
        if (c0 + 32 < 256) stage(c0 + 32, bi ^ 1);   // prefetch next chunk

        // ---- S = Q * K^T (two 16x16 N-tiles) ----
        int kb0 = hi ? 16 : 0;
        v16bf bk0 = frag16(&Ks[bi][m][kb0],      &Ks[bi][m][kb0 + 8]);
        v16bf bk1 = frag16(&Ks[bi][16 + m][kb0], &Ks[bi][16 + m][kb0 + 8]);
        v8f z = {0.f,0.f,0.f,0.f,0.f,0.f,0.f,0.f};
        v8f s0 = __builtin_amdgcn_wmma_f32_16x16x32_bf16(false, aq, false, bk0,
                                                         (short)0, z, false, false);
        v8f s1 = __builtin_amdgcn_wmma_f32_16x16x32_bf16(false, aq, false, bk1,
                                                         (short)0, z, false, false);

        // ---- online softmax (rows spread across 16-lane groups) ----
        for (int r = 0; r < 8; ++r) {
            float sv0 = s0[r] * scale, sv1 = s1[r] * scale;
            float cm = fmaxf(sv0, sv1);
            for (int off = 8; off >= 1; off >>= 1)
                cm = fmaxf(cm, __shfl_xor(cm, off, 32));
            float mn  = fmaxf(mrow[r], cm);
            float fac = __expf(mrow[r] - mn);
            float p0  = __expf(sv0 - mn);
            float p1  = __expf(sv1 - mn);
            float ps  = p0 + p1;
            for (int off = 8; off >= 1; off >>= 1)
                ps += __shfl_xor(ps, off, 32);
            lrow[r] = lrow[r] * fac + ps;
            mrow[r] = mn;
            o0[r] *= fac; o1[r] *= fac;
            int mm = r + 8 * hi;
            Ps[wv][mm][m]      = (__bf16)p0;
            Ps[wv][mm][16 + m] = (__bf16)p1;
        }
        // Ps is wave-private: intra-wave LDS dependency only.

        // ---- O += P * V (two 16-wide channel halves) ----
        v16bf ap  = frag16(&Ps[wv][m][hi ? 16 : 0], &Ps[wv][m][hi ? 24 : 8]);
        v16bf bv0 = frag16(&Vt[bi][m][kb0],      &Vt[bi][m][kb0 + 8]);
        v16bf bv1 = frag16(&Vt[bi][16 + m][kb0], &Vt[bi][16 + m][kb0 + 8]);
        o0 = __builtin_amdgcn_wmma_f32_16x16x32_bf16(false, ap, false, bv0,
                                                     (short)0, o0, false, false);
        o1 = __builtin_amdgcn_wmma_f32_16x16x32_bf16(false, ap, false, bv1,
                                                     (short)0, o1, false, false);

        wait_async0();        // own async stores into buf^1 complete
        __syncthreads();      // everyone done reading buf; next chunk visible
    }

    // normalize, bounce through LDS, write out (coalesced along tokens for dir0)
    for (int r = 0; r < 8; ++r) {
        float inv = 1.0f / lrow[r];
        int mm = r + 8 * hi;
        Ob[wv][m][mm]      = o0[r] * inv;
        Ob[wv][16 + m][mm] = o1[r] * inv;
    }
    size_t base = (size_t)(dir * 2 + b) * SPLIT * HWN
                + (size_t)choff * HWN + (dir ? so : so * 256);
    for (int e = lane; e < 512; e += 32) {
        int c = e >> 4, t = e & 15;
        out[base + (size_t)c * HWN + (size_t)(qt0 + t) * stride] = Ob[wv][c][t];
    }
}

// ---------------------------------------------------------------------------
extern "C" void kernel_launch(void* const* d_in, const int* in_sizes, int n_in,
                              void* d_out, int out_size, void* d_ws, size_t ws_size,
                              hipStream_t stream) {
    (void)in_sizes; (void)n_in; (void)out_size; (void)ws_size;
    const float* x      = (const float*)d_in[0];
    const float* ln_w   = (const float*)d_in[1];
    const float* ln_b   = (const float*)d_in[2];
    const float* conv_w = (const float*)d_in[3];
    const float* conv_b = (const float*)d_in[4];
    const float* w_qh = (const float*)d_in[5];   const float* b_qh = (const float*)d_in[6];
    const float* w_kh = (const float*)d_in[7];   const float* b_kh = (const float*)d_in[8];
    const float* w_vh = (const float*)d_in[9];   const float* b_vh = (const float*)d_in[10];
    const float* w_qv = (const float*)d_in[11];  const float* b_qv = (const float*)d_in[12];
    const float* w_kv = (const float*)d_in[13];  const float* b_kv = (const float*)d_in[14];
    const float* w_vv = (const float*)d_in[15];  const float* b_vv = (const float*)d_in[16];
    float* out = (float*)d_out;

    char* ws = (char*)d_ws;
    const size_t SZ_XC  = (size_t)PTOT * CHANNELS * sizeof(__bf16); // 80 MiB
    const size_t SZ_QKV = (size_t)PTOT * SPLIT    * sizeof(__bf16); // 40 MiB
    __bf16* xc = (__bf16*)ws;
    __bf16* xn = (__bf16*)(ws + SZ_XC);          // dead after conv; qkv overlaps it
    __bf16* qh = (__bf16*)(ws + SZ_XC + 0 * SZ_QKV);
    __bf16* kh = (__bf16*)(ws + SZ_XC + 1 * SZ_QKV);
    __bf16* vh = (__bf16*)(ws + SZ_XC + 2 * SZ_QKV);
    __bf16* qv = (__bf16*)(ws + SZ_XC + 3 * SZ_QKV);
    __bf16* kv = (__bf16*)(ws + SZ_XC + 4 * SZ_QKV);
    __bf16* vv = (__bf16*)(ws + SZ_XC + 5 * SZ_QKV);

    ln_kernel<<<PTOT / 256, 256, 0, stream>>>(x, ln_w, ln_b, xn);
    conv_gemm_kernel<<<dim3(PTOT / 16, CHANNELS / 64), 128, 0, stream>>>(xn, conv_w, conv_b, xc);

    proj_gemm_kernel<<<dim3(PTOT / 16, 2), 160, 0, stream>>>(xc, w_qh, b_qh, qh, 0);
    proj_gemm_kernel<<<dim3(PTOT / 16, 2), 160, 0, stream>>>(xc, w_kh, b_kh, kh, 0);
    proj_gemm_kernel<<<dim3(PTOT / 16, 2), 160, 0, stream>>>(xc, w_vh, b_vh, vh, 0);
    proj_gemm_kernel<<<dim3(PTOT / 16, 2), 160, 0, stream>>>(xc, w_qv, b_qv, qv, SPLIT);
    proj_gemm_kernel<<<dim3(PTOT / 16, 2), 160, 0, stream>>>(xc, w_kv, b_kv, kv, SPLIT);
    proj_gemm_kernel<<<dim3(PTOT / 16, 2), 160, 0, stream>>>(xc, w_vv, b_vv, vv, SPLIT);

    axial_attn_kernel<<<dim3(512, HEADS, 2), 256, 0, stream>>>(qh, kh, vh, out, 0);
    axial_attn_kernel<<<dim3(512, HEADS, 2), 256, 0, stream>>>(qv, kv, vv, out, 1);
}
